// ClassicalAttention_65481071408069
// MI455X (gfx1250) — compile-verified, matrix operand order checked by reading, not run on hardware
//
#include <hip/hip_runtime.h>

// ---------------------------------------------------------------------------
// CDNA5 (gfx1250) WMMA attention: out = softmax((x Wq^T)/sqrt(D) (x Wk^T)^T) (x Wv^T)
// B=4, S=2048, D=1024, fp32 in/out, bf16 matrix-core math with fp32 accum.
// Double-buffered LDS, K-step 64, async global->LDS where available.
// ---------------------------------------------------------------------------

typedef __attribute__((ext_vector_type(16))) __bf16 v16bf;
typedef __attribute__((ext_vector_type(8)))  float  v8f;
typedef __attribute__((ext_vector_type(4)))  int    v4i;

#define ATT_B 4
#define ATT_S 2048
#define ATT_D 1024

// LDS tile: 128 rows x 64 bf16 data + 16 bf16 pad -> row stride 80 bf16 (160B)
#define LDS_STRIDE 80

#define AS1 __attribute__((address_space(1)))
#define AS3 __attribute__((address_space(3)))

#if defined(__has_builtin)
#  if __has_builtin(__builtin_amdgcn_global_load_async_to_lds_b128) && \
      __has_builtin(__builtin_amdgcn_s_wait_asynccnt)
#    define ASYNC_LDS_AVAILABLE 1
#  endif
#endif
#ifndef ASYNC_LDS_AVAILABLE
#  define ASYNC_LDS_AVAILABLE 0
#endif

enum { MODE_F32 = 0, MODE_BF16 = 1, MODE_BF16_TRANS = 2 };

__device__ __forceinline__ unsigned short f32_to_bf16(float f) {
  unsigned int u = __float_as_uint(f);
  unsigned int r = u + 0x7fffu + ((u >> 16) & 1u);  // round to nearest even
  return (unsigned short)(r >> 16);
}

// Pack two f32 -> bf16x2 in one v_perm_b32 (round-half-up).
__device__ __forceinline__ unsigned int pack_bf16x2(float lo, float hi) {
  unsigned int ulo = __float_as_uint(lo) + 0x8000u;
  unsigned int uhi = __float_as_uint(hi) + 0x8000u;
  return __builtin_amdgcn_perm(uhi, ulo, 0x07060302u);
}

// Load one 16x32 bf16 fragment slice for this lane from an LDS tile.
// ISA 16-bit A/B layout: lanes 0-15 hold K = 0..7 then 16..23,
// lanes 16-31 hold K = 8..15 then 24..31 (relative to ksub*32).
__device__ __forceinline__ v16bf frag_load(const unsigned short* __restrict__ tile,
                                           int row, int half, int ksub) {
  union { v16bf v; uint4 q[2]; } u;
  const unsigned short* p = tile + row * LDS_STRIDE + ksub * 32 + half * 8;
  u.q[0] = *(const uint4*)(p);
  u.q[1] = *(const uint4*)(p + 16);
  return u.v;
}

// ---------------------------------------------------------------------------
// Generic tiled WMMA GEMM, C[M,N] = scale * A * op(B).
//   AMODE: MODE_F32 (A fp32, convert at staging) or MODE_BF16 (A bf16, NT)
//   BMODE: MODE_F32 / MODE_BF16 (B is N x K row-major, we use B^T)
//          MODE_BF16_TRANS      (B is K x N row-major, transposed at staging)
// TC: float or unsigned short (bf16 bits).
// Block: 256 threads = 8 wave32; block tile 128x128; wave tile 32x64;
// K-step 64; double-buffered LDS; grid = (N/128, M/128, batches).
// ---------------------------------------------------------------------------
template <int AMODE, int BMODE, typename TC>
__global__ __launch_bounds__(256)
void gemm_wmma(const void* __restrict__ Av, const void* __restrict__ Bv,
               TC* __restrict__ C, int K, int lda, int ldb, int ldc,
               long long sA, long long sB, long long sC, float scale) {
  __shared__ __align__(16) unsigned short Atile[2][128 * LDS_STRIDE];
  __shared__ __align__(16) unsigned short Btile[2][128 * LDS_STRIDE];

  const int t    = threadIdx.x;
  const int lane = t & 31;
  const int wave = t >> 5;
  const int wm   = wave & 3;   // 4 waves along M (32 rows each)
  const int wn   = wave >> 2;  // 2 waves along N (64 cols each)
  const int lrow = lane & 15;
  const int half = lane >> 4;

  const int row0 = blockIdx.y * 128;
  const int col0 = blockIdx.x * 128;

  const float*          Af = (const float*)Av + (size_t)blockIdx.z * sA;
  const unsigned short* Au = (const unsigned short*)Av + (size_t)blockIdx.z * sA;
  const float*          Bf = (const float*)Bv + (size_t)blockIdx.z * sB;
  const unsigned short* Bu = (const unsigned short*)Bv + (size_t)blockIdx.z * sB;
  TC*                   Cb = C + (size_t)blockIdx.z * sC;

  constexpr bool kAsyncA = (AMODE == MODE_BF16) && ASYNC_LDS_AVAILABLE;
  constexpr bool kAsyncB = (BMODE == MODE_BF16) && ASYNC_LDS_AVAILABLE;

  // NT staging coords: 2 threads per row, 32 elements each
  const int srow = t >> 1;
  const int scol = (t & 1) * 32;
  // BTRANS staging coords: 64 k-rows x 128 n-cols, 32 elements each
  const int bk = t >> 2;
  const int bn = (t & 3) * 32;

  v8f acc[2][4];
#pragma unroll
  for (int i = 0; i < 2; ++i)
#pragma unroll
    for (int j = 0; j < 4; ++j)
#pragma unroll
      for (int e = 0; e < 8; ++e) acc[i][j][e] = 0.0f;

  float4 aRegF[8]; uint4 aRegU[4];
  float4 bRegF[8]; uint4 bRegU[4];

  auto issueA = [&](int kk, int nbuf) {
    if constexpr (AMODE == MODE_F32) {
      const float* src = Af + (size_t)(row0 + srow) * lda + kk + scol;
#pragma unroll
      for (int i = 0; i < 8; ++i) aRegF[i] = ((const float4*)src)[i];
    } else {
      const unsigned short* src = Au + (size_t)(row0 + srow) * lda + kk + scol;
#if ASYNC_LDS_AVAILABLE
      if constexpr (kAsyncA) {
        unsigned short* dst = &Atile[nbuf][srow * LDS_STRIDE + scol];
        __builtin_amdgcn_global_load_async_to_lds_b128((AS1 v4i*)src, (AS3 v4i*)dst, 0, 0);
        __builtin_amdgcn_global_load_async_to_lds_b128((AS1 v4i*)src, (AS3 v4i*)dst, 16, 0);
        __builtin_amdgcn_global_load_async_to_lds_b128((AS1 v4i*)src, (AS3 v4i*)dst, 32, 0);
        __builtin_amdgcn_global_load_async_to_lds_b128((AS1 v4i*)src, (AS3 v4i*)dst, 48, 0);
      } else
#endif
      {
#pragma unroll
        for (int i = 0; i < 4; ++i) aRegU[i] = ((const uint4*)src)[i];
      }
    }
    (void)nbuf;
  };

  auto storeA = [&](int nbuf) {
    if constexpr (AMODE == MODE_F32) {
      uint4* dst = (uint4*)&Atile[nbuf][srow * LDS_STRIDE + scol];
#pragma unroll
      for (int i = 0; i < 4; ++i) {
        uint4 o;
        o.x = pack_bf16x2(aRegF[2 * i].x, aRegF[2 * i].y);
        o.y = pack_bf16x2(aRegF[2 * i].z, aRegF[2 * i].w);
        o.z = pack_bf16x2(aRegF[2 * i + 1].x, aRegF[2 * i + 1].y);
        o.w = pack_bf16x2(aRegF[2 * i + 1].z, aRegF[2 * i + 1].w);
        dst[i] = o;
      }
    } else if constexpr (!kAsyncA) {
      uint4* dst = (uint4*)&Atile[nbuf][srow * LDS_STRIDE + scol];
#pragma unroll
      for (int i = 0; i < 4; ++i) dst[i] = aRegU[i];
    } else {
      (void)nbuf;
    }
  };

  auto issueB = [&](int kk, int nbuf) {
    if constexpr (BMODE == MODE_F32) {
      const float* src = Bf + (size_t)(col0 + srow) * ldb + kk + scol;
#pragma unroll
      for (int i = 0; i < 8; ++i) bRegF[i] = ((const float4*)src)[i];
    } else if constexpr (BMODE == MODE_BF16) {
      const unsigned short* src = Bu + (size_t)(col0 + srow) * ldb + kk + scol;
#if ASYNC_LDS_AVAILABLE
      if constexpr (kAsyncB) {
        unsigned short* dst = &Btile[nbuf][srow * LDS_STRIDE + scol];
        __builtin_amdgcn_global_load_async_to_lds_b128((AS1 v4i*)src, (AS3 v4i*)dst, 0, 0);
        __builtin_amdgcn_global_load_async_to_lds_b128((AS1 v4i*)src, (AS3 v4i*)dst, 16, 0);
        __builtin_amdgcn_global_load_async_to_lds_b128((AS1 v4i*)src, (AS3 v4i*)dst, 32, 0);
        __builtin_amdgcn_global_load_async_to_lds_b128((AS1 v4i*)src, (AS3 v4i*)dst, 48, 0);
      } else
#endif
      {
#pragma unroll
        for (int i = 0; i < 4; ++i) bRegU[i] = ((const uint4*)src)[i];
      }
    } else {  // MODE_BF16_TRANS: B is K x N row-major, read 64-row k slab
      const unsigned short* src = Bu + (size_t)(kk + bk) * ldb + col0 + bn;
#pragma unroll
      for (int i = 0; i < 4; ++i) bRegU[i] = ((const uint4*)src)[i];
    }
    (void)nbuf;
  };

  auto storeB = [&](int nbuf) {
    if constexpr (BMODE == MODE_F32) {
      uint4* dst = (uint4*)&Btile[nbuf][srow * LDS_STRIDE + scol];
#pragma unroll
      for (int i = 0; i < 4; ++i) {
        uint4 o;
        o.x = pack_bf16x2(bRegF[2 * i].x, bRegF[2 * i].y);
        o.y = pack_bf16x2(bRegF[2 * i].z, bRegF[2 * i].w);
        o.z = pack_bf16x2(bRegF[2 * i + 1].x, bRegF[2 * i + 1].y);
        o.w = pack_bf16x2(bRegF[2 * i + 1].z, bRegF[2 * i + 1].w);
        dst[i] = o;
      }
    } else if constexpr (BMODE == MODE_BF16) {
      if constexpr (!kAsyncB) {
        uint4* dst = (uint4*)&Btile[nbuf][srow * LDS_STRIDE + scol];
#pragma unroll
        for (int i = 0; i < 4; ++i) dst[i] = bRegU[i];
      } else {
        (void)nbuf;
      }
    } else {  // transpose scatter: Btile[n][k]
      union { uint4 q[4]; unsigned short h[32]; } u;
#pragma unroll
      for (int i = 0; i < 4; ++i) u.q[i] = bRegU[i];
#pragma unroll
      for (int e = 0; e < 32; ++e)
        Btile[nbuf][(bn + e) * LDS_STRIDE + bk] = u.h[e];
    }
  };

  auto compute = [&](int cbuf) {
#pragma unroll
    for (int ksub = 0; ksub < 2; ++ksub) {
      v16bf aF[2], bF[4];
#pragma unroll
      for (int i = 0; i < 2; ++i)
        aF[i] = frag_load(&Atile[cbuf][0], wm * 32 + i * 16 + lrow, half, ksub);
#pragma unroll
      for (int j = 0; j < 4; ++j)
        bF[j] = frag_load(&Btile[cbuf][0], wn * 64 + j * 16 + lrow, half, ksub);
#pragma unroll
      for (int i = 0; i < 2; ++i)
#pragma unroll
        for (int j = 0; j < 4; ++j)
          acc[i][j] = __builtin_amdgcn_wmma_f32_16x16x32_bf16(
              false, aF[i], false, bF[j], (short)0, acc[i][j], false, false);
    }
  };

  // ---- prologue: stage tile 0 ----
  issueA(0, 0);
  issueB(0, 0);
  storeA(0);
  storeB(0);
#if ASYNC_LDS_AVAILABLE
  if constexpr (kAsyncA || kAsyncB) __builtin_amdgcn_s_wait_asynccnt(0);
#endif
  __syncthreads();

  // ---- main loop: one barrier per K-step of 64 ----
  int buf = 0;
  for (int kk = 0; kk < K; kk += 64) {
    const bool hasNext = (kk + 64) < K;
    if (hasNext) {
      issueA(kk + 64, buf ^ 1);   // global loads (or async DMA) for next tile
      issueB(kk + 64, buf ^ 1);
    }
    compute(buf);                 // 16 WMMAs behind the in-flight loads
    if (hasNext) {
      storeA(buf ^ 1);            // convert + ds_store (no-op for async path)
      storeB(buf ^ 1);
#if ASYNC_LDS_AVAILABLE
      if constexpr (kAsyncA || kAsyncB) __builtin_amdgcn_s_wait_asynccnt(0);
#endif
    }
    __syncthreads();
    buf ^= 1;
  }

  // ---- epilogue: C/D layout VGPR r -> row half*8+r, col = lane&15 ----
#pragma unroll
  for (int i = 0; i < 2; ++i) {
#pragma unroll
    for (int j = 0; j < 4; ++j) {
      const int gr0 = row0 + wm * 32 + i * 16 + half * 8;
      const int gc  = col0 + wn * 64 + j * 16 + lrow;
#pragma unroll
      for (int r = 0; r < 8; ++r) {
        float val = scale * acc[i][j][r];
        size_t idx = (size_t)(gr0 + r) * ldc + gc;
        if constexpr (sizeof(TC) == 2)
          Cb[idx] = (TC)f32_to_bf16(val);
        else
          Cb[idx] = (TC)val;
      }
    }
  }
}

// ---------------------------------------------------------------------------
// In-place row softmax for 2048-wide fp32 rows (one 256-thread block per row).
// ---------------------------------------------------------------------------
__global__ __launch_bounds__(256)
void softmax_rows_2048(float* __restrict__ S) {
  __shared__ float red[256];
  const int t = threadIdx.x;
  float* row = S + (size_t)blockIdx.x * 2048;

  float4 a = ((const float4*)(row + t * 8))[0];
  float4 b = ((const float4*)(row + t * 8))[1];
  float v[8] = {a.x, a.y, a.z, a.w, b.x, b.y, b.z, b.w};

  float mx = v[0];
#pragma unroll
  for (int e = 1; e < 8; ++e) mx = fmaxf(mx, v[e]);
  red[t] = mx;
  __syncthreads();
  for (int s = 128; s > 0; s >>= 1) {
    if (t < s) red[t] = fmaxf(red[t], red[t + s]);
    __syncthreads();
  }
  mx = red[0];
  __syncthreads();

  float sum = 0.0f;
#pragma unroll
  for (int e = 0; e < 8; ++e) {
    v[e] = __expf(v[e] - mx);
    sum += v[e];
  }
  red[t] = sum;
  __syncthreads();
  for (int s = 128; s > 0; s >>= 1) {
    if (t < s) red[t] += red[t + s];
    __syncthreads();
  }
  const float inv = 1.0f / red[0];

  float4 oa, ob;
  oa.x = v[0] * inv; oa.y = v[1] * inv; oa.z = v[2] * inv; oa.w = v[3] * inv;
  ob.x = v[4] * inv; ob.y = v[5] * inv; ob.z = v[6] * inv; ob.w = v[7] * inv;
  ((float4*)(row + t * 8))[0] = oa;
  ((float4*)(row + t * 8))[1] = ob;
}

// ---------------------------------------------------------------------------
// Host-side orchestration.
// d_in order: inputs, rotation_params (unused), entangle_params (unused),
//             w_q, w_k, w_v
// ---------------------------------------------------------------------------
extern "C" void kernel_launch(void* const* d_in, const int* in_sizes, int n_in,
                              void* d_out, int out_size, void* d_ws, size_t ws_size,
                              hipStream_t stream) {
  (void)in_sizes; (void)n_in; (void)out_size; (void)ws_size;

  const float* x  = (const float*)d_in[0];
  const float* wq = (const float*)d_in[3];
  const float* wk = (const float*)d_in[4];
  const float* wv = (const float*)d_in[5];
  float* out = (float*)d_out;

  const long long BS = (long long)ATT_B * ATT_S;  // 8192
  const long long SD = (long long)ATT_S * ATT_D;  // per-batch Q/K/V stride
  const long long SS = (long long)ATT_S * ATT_S;  // per-batch score stride

  // workspace: Q,K,V as bf16 bits (16 MB each), scores fp32 (67 MB, L2-resident)
  char* ws = (char*)d_ws;
  unsigned short* Qb = (unsigned short*)ws;
  unsigned short* Kb = Qb + BS * ATT_D;
  unsigned short* Vb = Kb + BS * ATT_D;
  float* Sc = (float*)(ws + 3ll * BS * ATT_D * 2);

  dim3 blk(256);

  // 1) QKV projections: C[8192,1024] = x[8192,1024] * W[1024,1024]^T
  dim3 gProj(ATT_D / 128, (unsigned)(BS / 128), 1);
  gemm_wmma<MODE_F32, MODE_F32, unsigned short><<<gProj, blk, 0, stream>>>(
      x, wq, Qb, ATT_D, ATT_D, ATT_D, ATT_D, 0, 0, 0, 1.0f / 32.0f);
  gemm_wmma<MODE_F32, MODE_F32, unsigned short><<<gProj, blk, 0, stream>>>(
      x, wk, Kb, ATT_D, ATT_D, ATT_D, ATT_D, 0, 0, 0, 1.0f);
  gemm_wmma<MODE_F32, MODE_F32, unsigned short><<<gProj, blk, 0, stream>>>(
      x, wv, Vb, ATT_D, ATT_D, ATT_D, ATT_D, 0, 0, 0, 1.0f);

  // 2) scores[b] = Q[b] * K[b]^T  (bf16 inputs -> async LDS path; fp32 out)
  dim3 gScore(ATT_S / 128, ATT_S / 128, ATT_B);
  gemm_wmma<MODE_BF16, MODE_BF16, float><<<gScore, blk, 0, stream>>>(
      Qb, Kb, Sc, ATT_D, ATT_D, ATT_D, ATT_S, SD, SD, SS, 1.0f);

  // 3) in-place softmax over each of B*S rows of length S
  softmax_rows_2048<<<dim3((unsigned)BS), blk, 0, stream>>>(Sc);

  // 4) out[b] = P[b] * V[b]  (P fp32 -> bf16 at staging, V staged transposed)
  dim3 gOut(ATT_D / 128, ATT_S / 128, ATT_B);
  gemm_wmma<MODE_F32, MODE_BF16_TRANS, float><<<gOut, blk, 0, stream>>>(
      Sc, Vb, out, ATT_S, ATT_S, ATT_D, ATT_D, SS, SD, SD, 1.0f);
}
